// MaskedPiecewiseLinearAutoregressiveTransform_91319594647696
// MI455X (gfx1250) — matrix operand order repaired and checked by reading, not compiled, not run on hardware
//
#include <hip/hip_runtime.h>
#include <hip/hip_bf16.h>

// MI455X / gfx1250, wave32. bf16 WMMA (f32 accumulate), fragment-packed weights,
// fused spline epilogue so the [8192 x 8192] params tensor never touches HBM.

typedef __attribute__((ext_vector_type(16))) __bf16 v16bf;
typedef __attribute__((ext_vector_type(8)))  __bf16 v8bf;
typedef __attribute__((ext_vector_type(8)))  float  v8f;

#define Df 64
#define Hf 512
#define Kb 128
#define Bn 8192

union Frag16 { v16bf v; v8bf h[2]; };

// Packed-weight index: for n-tile (16 rows) x k-step (32 deep), the 32 lanes'
// 16-element fragments are contiguous -> one 1KB contiguous burst per wave.
// lane = (n&15) | (((k>>4)&1)<<4), element = k&15.
__device__ __forceinline__ size_t wpack(int n, int k, int Kd) {
  return (size_t)((((n >> 4) * (Kd >> 5) + (k >> 5)) * 32 +
                   ((n & 15) | (((k >> 4) & 1) << 4))) * 16 + (k & 15));
}

// ---------------- elementwise prep kernels (mask + f32->bf16 + pack) --------

__global__ void k_cvt_in(const float* __restrict__ x, __bf16* __restrict__ o, int n) {
  int i = blockIdx.x * 256 + threadIdx.x;
  if (i < n) o[i] = (__bf16)x[i];
}

__global__ void k_mask_init(const float* __restrict__ w, __bf16* __restrict__ o) {
  int i = blockIdx.x * 256 + threadIdx.x;            // Hf*Df
  if (i < Hf * Df) {
    int n = i >> 6, k = i & 63;                      // row h, col d
    o[wpack(n, k, Df)] = (__bf16)(((n % 63) >= k) ? w[i] : 0.f);
  }
}

__global__ void k_mask_hid(const float* __restrict__ w, __bf16* __restrict__ o) {
  int i = blockIdx.x * 256 + threadIdx.x;            // 4*Hf*Hf
  if (i < 4 * Hf * Hf) {
    int l = i >> 18;                                 // which of 4 matrices
    int loc = i & (Hf * Hf - 1);
    int n = loc >> 9, k = loc & 511;                 // row h, col j
    o[(size_t)l * Hf * Hf + wpack(n, k, Hf)] =
        (__bf16)(((n % 63) >= (k % 63)) ? w[i] : 0.f);
  }
}

__global__ void k_mask_out(const float* __restrict__ w, __bf16* __restrict__ o) {
  int i = blockIdx.x * 256 + threadIdx.x;            // Df*Kb*Hf
  if (i < Df * Kb * Hf) {
    int n = i >> 9;        // output row (d*Kb + bin)
    int k = i & 511;       // hidden col
    int dF = n >> 7;       // feature index
    o[wpack(n, k, Hf)] = (__bf16)((dF > (k % 63)) ? w[i] : 0.f);
  }
}

__global__ void k_zero(float* __restrict__ p, int n) {
  int i = blockIdx.x * 256 + threadIdx.x;
  if (i < n) p[i] = 0.f;
}

// ---------------- fused WMMA GEMM (+ optional spline epilogue) ----------------
// Workgroup tile: 128 rows x 128 cols. 8 waves in a 2(M) x 4(N) grid; each wave
// computes 4 m-subtiles x 2 n-subtiles => 8 wmma per 32-K step with
// 8 ds_load_b128 (A) + 4 coalesced global_load_b128 (packed B).

__global__ __launch_bounds__(256) void gemm_made(
    const __bf16* __restrict__ A,     // [Bn, Kd] bf16 row-major
    const __bf16* __restrict__ W,     // [Ntot, Kd] bf16, fragment-packed
    const float*  __restrict__ bias,  // [Ntot]
    int Kd,
    const float*  __restrict__ resid, // [Bn, Hf] or null
    float*        __restrict__ tOut,  // [Bn, Hf] or null
    __bf16*       __restrict__ actOut,// [Bn, Hf] or null
    int reluFlag,
    int doSpline,
    const float*  __restrict__ xIn,   // [Bn, Df] (spline only)
    float*        __restrict__ dOut)  // [Bn*Df] outputs + [Bn] logabsdet
{
  __shared__ __align__(16) char smem[35072];
  __bf16* As = (__bf16*)smem;                       // [128][72] bf16 (GEMM phase)
  float (*P)[Kb + 1] = (float (*)[Kb + 1])smem;     // [64][129] f32 (spline phase)
  float* red  = (float*)(smem + 33024);             // [64][4]
  float* red2 = red + 256;                          // [64][4]

  const int tid  = threadIdx.x;
  const int lane = tid & 31;
  const int w    = tid >> 5;       // wave 0..7
  const int wm   = w >> 2;         // wave m-group 0..1 (64 rows each)
  const int wn   = w & 3;          // wave n-group 0..3 (32 cols each)
  const int g    = lane >> 4;      // half-wave group
  const int ln   = lane & 15;
  const int b0   = blockIdx.x * 128;
  const int nBlk = blockIdx.y;

  v8f acc[4][2] = {};

  const int kSteps = Kd >> 5;
  const int nTile0 = nBlk * 8 + wn * 2;   // global 16-wide n-tile index (nt=0)

  for (int kc = 0; kc < Kd; kc += 64) {
    // cooperative stage of A[128 x 64] chunk into LDS (16B per transfer)
#pragma unroll
    for (int it = 0; it < 4; ++it) {
      int cid = tid * 4 + it;          // 0..1023
      int r   = cid >> 3;
      int cg  = cid & 7;
      *(v8bf*)(As + r * 72 + cg * 8) =
          *(const v8bf*)(A + (size_t)(b0 + r) * Kd + kc + cg * 8);
    }
    __syncthreads();
#pragma unroll
    for (int ks = 0; ks < 64; ks += 32) {
      const int kstep = (kc + ks) >> 5;
      // packed B fragments: one contiguous 1KB burst per wave per n-tile
      Frag16 bf[2];
#pragma unroll
      for (int nt = 0; nt < 2; ++nt) {
        const v8bf* bp = (const v8bf*)(W +
            (((size_t)(nTile0 + nt) * kSteps + kstep) * 32 + lane) * 16);
        bf[nt].h[0] = bp[0];
        bf[nt].h[1] = bp[1];
      }
#pragma unroll
      for (int mt = 0; mt < 4; ++mt) {
        // A fragment: lane holds row wm*64+mt*16+ln, K = e+16g
        Frag16 af;
        const __bf16* ar = As + (wm * 64 + mt * 16 + ln) * 72;
        af.h[0] = *(const v8bf*)(ar + ks + 16 * g);
        af.h[1] = *(const v8bf*)(ar + ks + 16 * g + 8);
#pragma unroll
        for (int nt = 0; nt < 2; ++nt)
          acc[mt][nt] = __builtin_amdgcn_wmma_f32_16x16x32_bf16(
              false, af.v, false, bf[nt].v, (short)0, acc[mt][nt], false, false);
      }
    }
    __syncthreads();
  }

  float bv[2];
#pragma unroll
  for (int nt = 0; nt < 2; ++nt)
    bv[nt] = bias[nBlk * 128 + wn * 32 + nt * 16 + ln];

  if (!doSpline) {
#pragma unroll
    for (int nt = 0; nt < 2; ++nt) {
      const int nG = nBlk * 128 + wn * 32 + nt * 16 + ln;
#pragma unroll
      for (int mt = 0; mt < 4; ++mt) {
#pragma unroll
        for (int v = 0; v < 8; ++v) {
          int r = b0 + wm * 64 + mt * 16 + 8 * g + v;   // C/D layout: M = v + 8g
          float val = acc[mt][nt][v] + bv[nt];
          if (resid)  val += resid[(size_t)r * Hf + nG];
          if (tOut)   tOut[(size_t)r * Hf + nG] = val;
          if (actOut) actOut[(size_t)r * Hf + nG] =
              (__bf16)(reluFlag ? fmaxf(val, 0.f) : val);
        }
      }
    }
    return;
  }

  // ---- fused piecewise-linear spline, two 64-row passes through LDS ----
  const int d = nBlk;                 // 128 cols == the K bins of one feature
  for (int half = 0; half < 2; ++half) {
    __syncthreads();                  // P free (aliases As / previous pass)
    if (wm == half) {
#pragma unroll
      for (int nt = 0; nt < 2; ++nt)
#pragma unroll
        for (int mt = 0; mt < 4; ++mt)
#pragma unroll
          for (int v = 0; v < 8; ++v)
            P[mt * 16 + 8 * g + v][wn * 32 + nt * 16 + ln] =
                acc[mt][nt][v] + bv[nt];
    }
    __syncthreads();

    const int r  = tid >> 2;          // row within pass, 0..63
    const int c  = tid & 3;           // 32-bin chunk
    const int cb = c * 32;
    const int rg = b0 + half * 64 + r;

    float mx = -3.0e38f;
    for (int i = 0; i < 32; ++i) mx = fmaxf(mx, P[r][cb + i]);
    red[r * 4 + c] = mx;
    __syncthreads();
    float rowmax = fmaxf(fmaxf(red[r * 4], red[r * 4 + 1]),
                         fmaxf(red[r * 4 + 2], red[r * 4 + 3]));
    float s = 0.f;
    for (int i = 0; i < 32; ++i) s += __expf(P[r][cb + i] - rowmax);
    red2[r * 4 + c] = s;
    __syncthreads();
    float Z = red2[r * 4] + red2[r * 4 + 1] + red2[r * 4 + 2] + red2[r * 4 + 3];

    float x   = xIn[(size_t)rg * Df + d];
    float pos = x * (float)Kb;
    int   idx = (int)floorf(pos);
    idx = idx < 0 ? 0 : (idx > Kb - 1 ? Kb - 1 : idx);
    float alpha = pos - (float)idx;

    if ((idx >> 5) == c) {
      float left = 0.f;
      for (int cc = 0; cc < c; ++cc) left += red2[r * 4 + cc];
      for (int i = cb; i < idx; ++i) left += __expf(P[r][i] - rowmax);
      float vi  = P[r][idx];
      float pdf = __expf(vi - rowmax) / Z;
      float outv = left / Z + alpha * pdf;
      outv = outv < 0.f ? 0.f : (outv > 1.f ? 1.f : outv);
      dOut[(size_t)rg * Df + d] = outv;
      float lad = (vi - rowmax) - __logf(Z) + 4.852030263919617f;  // + ln(128)
      unsafeAtomicAdd(dOut + (size_t)Bn * Df + rg, lad);
    }
  }
}

// ---------------- host launch ----------------

extern "C" void kernel_launch(void* const* d_in, const int* in_sizes, int n_in,
                              void* d_out, int out_size, void* d_ws, size_t ws_size,
                              hipStream_t stream) {
  const float* x  = (const float*)d_in[0];   // [Bn, Df]
  const float* Wi = (const float*)d_in[1];   // [Hf, Df]
  const float* bi = (const float*)d_in[2];   // [Hf]
  const float* Wb = (const float*)d_in[3];   // [2,2,Hf,Hf]
  const float* bb = (const float*)d_in[4];   // [2,2,Hf]
  const float* Wo = (const float*)d_in[5];   // [Df*Kb, Hf]
  const float* bo = (const float*)d_in[6];   // [Df*Kb]
  float* out = (float*)d_out;

  char* ws = (char*)d_ws;
  size_t o = 0;
  auto alloc = [&](size_t bytes) -> void* {
    void* p = ws + o;
    o = (o + bytes + 255) & ~(size_t)255;
    return p;
  };
  __bf16* Xb  = (__bf16*)alloc((size_t)Bn * Df * 2);
  __bf16* Wib = (__bf16*)alloc((size_t)Hf * Df * 2);
  __bf16* Wbb = (__bf16*)alloc((size_t)4 * Hf * Hf * 2);
  __bf16* Wob = (__bf16*)alloc((size_t)Df * Kb * Hf * 2);
  float*  t   = (float*) alloc((size_t)Bn * Hf * 4);
  __bf16* tb  = (__bf16*)alloc((size_t)Bn * Hf * 2);
  __bf16* hb  = (__bf16*)alloc((size_t)Bn * Hf * 2);

  dim3 blk(256);
  auto nb = [](int n) { return (n + 255) / 256; };

  // weight masking + f32 -> bf16 + fragment packing
  k_cvt_in   <<<nb(Bn * Df), blk, 0, stream>>>(x, Xb, Bn * Df);
  k_mask_init<<<nb(Hf * Df), blk, 0, stream>>>(Wi, Wib);
  k_mask_hid <<<nb(4 * Hf * Hf), blk, 0, stream>>>(Wb, Wbb);
  k_mask_out <<<nb(Df * Kb * Hf), blk, 0, stream>>>(Wo, Wob);
  k_zero     <<<nb(Bn), blk, 0, stream>>>(out + (size_t)Bn * Df, Bn);

  dim3 gH(Bn / 128, Hf / 128);     // (64, 4)
  dim3 gO(Bn / 128, Df);           // (64, 64)

  // t = X @ (Wi*m_in)^T + bi ; tb = bf16(relu(t))
  gemm_made<<<gH, blk, 0, stream>>>(Xb, Wib, bi, Df,
                                    nullptr, t, tb, 1, 0, nullptr, nullptr);
  // block 0
  gemm_made<<<gH, blk, 0, stream>>>(tb, Wbb + (size_t)0 * Hf * Hf, bb + 0 * Hf, Hf,
                                    nullptr, nullptr, hb, 1, 0, nullptr, nullptr);
  gemm_made<<<gH, blk, 0, stream>>>(hb, Wbb + (size_t)1 * Hf * Hf, bb + 1 * Hf, Hf,
                                    t, t, tb, 1, 0, nullptr, nullptr);
  // block 1
  gemm_made<<<gH, blk, 0, stream>>>(tb, Wbb + (size_t)2 * Hf * Hf, bb + 2 * Hf, Hf,
                                    nullptr, nullptr, hb, 1, 0, nullptr, nullptr);
  gemm_made<<<gH, blk, 0, stream>>>(hb, Wbb + (size_t)3 * Hf * Hf, bb + 3 * Hf, Hf,
                                    t, t, tb, /*relu=*/0, 0, nullptr, nullptr);
  // final projection fused with spline (params stay in LDS)
  gemm_made<<<gO, blk, 0, stream>>>(tb, Wob, bo, Hf,
                                    nullptr, nullptr, nullptr, 0, 1, x, out);

  (void)in_sizes; (void)n_in; (void)out_size; (void)ws_size;
}
